// TemporalAttention_83726092468281
// MI455X (gfx1250) — compile-verified, matrix operand order checked by reading, not run on hardware
//
#include <hip/hip_runtime.h>
#include <hip/hip_bf16.h>

// MI455X / gfx1250, wave32, WMMA bf16 path.
//
// Pipeline:
//   k0: f32 -> bf16 staging (x, Wqkv, Wout)
//   k1: QKV GEMM (bf16 WMMA, f32 acc) + fused RoPE; V stored transposed
//   k2: causal flash attention (QK^T + online softmax + PV), P transposed via LDS
//   k3: output GEMM -> f32 d_out

typedef __bf16 bf16_t;
typedef __attribute__((ext_vector_type(16))) __bf16 v16bf;
typedef __attribute__((ext_vector_type(8)))  __bf16 v8bf;
typedef __attribute__((ext_vector_type(8)))  float  v8f;

#define BB 256
#define TT 256
#define DD 512
#define HH 8
#define HDD 64

union FragU { v16bf v; v8bf h[2]; };

// Load a 16x32 bf16 fragment (A-layout, also used for B since it mirrors with
// lane=N) from a row-major source: lane r<16 holds row r; VGPR halves hold
// K = k0+[0..7] (lanes 0-15) / k0+[8..15] (lanes 16-31) and +16 for upper 8.
__device__ __forceinline__ v16bf load_frag16x32(const bf16_t* __restrict__ tile_base,
                                                int stride, int k0) {
  const int lane = threadIdx.x & 31;
  const int r    = lane & 15;
  const int kk   = k0 + ((lane >> 4) << 3);
  const bf16_t* p = tile_base + r * stride + kk;
  FragU u;
  u.h[0] = *(const v8bf*)(p);
  u.h[1] = *(const v8bf*)(p + 16);
  return u.v;
}

__device__ __forceinline__ v8f wmma_bf16(v16bf a, v16bf b, v8f c) {
  // v_wmma_f32_16x16x32_bf16
  return __builtin_amdgcn_wmma_f32_16x16x32_bf16(false, a, false, b, (short)0, c,
                                                 false, false);
}

// ---------------------------------------------------------------- k0: convert
__global__ void cvt_f32_bf16_kernel(const float* __restrict__ src,
                                    bf16_t* __restrict__ dst, int n) {
  int i = blockIdx.x * blockDim.x + threadIdx.x;
  const int stride = gridDim.x * blockDim.x;
  for (; i < n; i += stride) dst[i] = (bf16_t)src[i];
}

// ------------------------------------------------------- k1: QKV GEMM + RoPE
// M = B*T = 65536, N = 3*D = 1536, K = 512.
// Block: 256 threads = 8 waves (4 along M x 2 along N); wave tile 64x32.
__global__ __launch_bounds__(256, 2)
void qkv_rope_kernel(const bf16_t* __restrict__ Xb,
                     const bf16_t* __restrict__ Wb,
                     const float* __restrict__ cosT,
                     const float* __restrict__ sinT,
                     bf16_t* __restrict__ Qb,
                     bf16_t* __restrict__ Kb,
                     bf16_t* __restrict__ Vt) {
  const int w    = threadIdx.x >> 5;
  const int wm   = w & 3;
  const int wn   = w >> 2;
  const int m0   = blockIdx.x * 256 + wm * 64;
  const int n0   = blockIdx.y * 64  + wn * 32;
  const int lane = threadIdx.x & 31;
  const int half = lane >> 4;
  const int ln   = lane & 15;

  v8f acc[4][2] = {};

  for (int k0 = 0; k0 < DD; k0 += 32) {
    __builtin_prefetch(Xb + m0 * DD + ((k0 + 32) & (DD - 1)), 0, 3);
    __builtin_prefetch(Wb + n0 * DD + ((k0 + 32) & (DD - 1)), 0, 3);
    v16bf a[4], b[2];
#pragma unroll
    for (int i = 0; i < 4; ++i)
      a[i] = load_frag16x32(Xb + (m0 + i * 16) * DD, DD, k0);
#pragma unroll
    for (int j = 0; j < 2; ++j)
      b[j] = load_frag16x32(Wb + (n0 + j * 16) * DD, DD, k0);
#pragma unroll
    for (int i = 0; i < 4; ++i)
#pragma unroll
      for (int j = 0; j < 2; ++j)
        acc[i][j] = wmma_bf16(a[i], b[j], acc[i][j]);
  }

  // Epilogue: RoPE on Q/K, scatter to [b][h][t][hd]; V scattered transposed.
#pragma unroll
  for (int i = 0; i < 4; ++i) {
#pragma unroll
    for (int j = 0; j < 2; ++j) {
      const int nb = n0 + j * 16 + ln;          // global column in [0,1536)
      const int region = nb >> 9;               // 0=Q 1=K 2=V
      const int nin = nb & (DD - 1);
      const int h  = nin >> 6;
      const int hd = nin & (HDD - 1);
#pragma unroll
      for (int r = 0; r < 8; ++r) {
        const int m = m0 + i * 16 + r + half * 8;
        const int b = m >> 8;                   // / TT
        const int t = m & (TT - 1);
        float val = acc[i][j][r];
        // RoPE pair partner lives in the adjacent lane (column n^1).
        const float partner = __shfl_xor(val, 1, 32);
        if (region < 2) {
          const float c = cosT[t * HDD + hd];
          const float s = sinT[t * HDD + hd];
          const float rot = (hd & 1) ? partner : -partner;
          val = val * c + rot * s;
          bf16_t* dst = (region == 0) ? Qb : Kb;
          dst[((b * HH + h) * TT + t) * HDD + hd] = (bf16_t)val;
        } else {
          // V transposed: [b][h][hd][t] so PV B-fragments load contiguously.
          Vt[((b * HH + h) * HDD + hd) * TT + t] = (bf16_t)val;
        }
      }
    }
  }
}

// ----------------------------------------------------- k2: causal attention
// One block (4 waves, 128 threads) per (b,h, 64-row query block).
// Each wave: 16 query rows, full-HD output accumulator (4 tiles), online
// softmax, P transposed C-layout -> A-layout through LDS.
__global__ __launch_bounds__(128, 4)
void attn_kernel(const bf16_t* __restrict__ Qb,
                 const bf16_t* __restrict__ Kb,
                 const bf16_t* __restrict__ Vt,
                 bf16_t* __restrict__ AO) {
  __shared__ __align__(16) bf16_t smem[4][16][32];

  const int w    = threadIdx.x >> 5;
  const int lane = threadIdx.x & 31;
  const int half = lane >> 4;
  const int ln   = lane & 15;

  const int bh = blockIdx.x >> 2;
  const int q0 = (blockIdx.x & 3) * 64;
  const int qw = q0 + w * 16;                      // wave's first query row

  const bf16_t* Qbase = Qb + (bh * TT + qw) * HDD;
  const bf16_t* Kbase = Kb + bh * TT * HDD;
  const bf16_t* Vbase = Vt + bh * HDD * TT;

  v16bf qf[2];
  qf[0] = load_frag16x32(Qbase, HDD, 0);
  qf[1] = load_frag16x32(Qbase, HDD, 32);

  v8f o[4] = {};
  float mrow[8], lrow[8];
#pragma unroll
  for (int r = 0; r < 8; ++r) { mrow[r] = -1e30f; lrow[r] = 0.0f; }

  const int kend = q0 + 64;                        // uniform across the block
  for (int kb = 0; kb < kend; kb += 32) {
    // S = Q K^T for a 16x32 key slab (two 16x16 N-tiles).
    v8f s0 = {}, s1 = {};
#pragma unroll
    for (int kc = 0; kc < 2; ++kc) {
      v16bf kf0 = load_frag16x32(Kbase + (kb +  0) * HDD, HDD, kc * 32);
      v16bf kf1 = load_frag16x32(Kbase + (kb + 16) * HDD, HDD, kc * 32);
      s0 = wmma_bf16(qf[kc], kf0, s0);
      s1 = wmma_bf16(qf[kc], kf1, s1);
    }

    // Scale + causal mask + online softmax update; write P tile to LDS.
#pragma unroll
    for (int r = 0; r < 8; ++r) {
      const int row  = qw + r + half * 8;
      const int key0 = kb + ln;
      const int key1 = kb + 16 + ln;
      float a0 = s0[r] * 0.125f;
      float a1 = s1[r] * 0.125f;
      a0 = (key0 <= row) ? a0 : -1e30f;
      a1 = (key1 <= row) ? a1 : -1e30f;

      float rmax = fmaxf(a0, a1);
#pragma unroll
      for (int msk = 1; msk <= 8; msk <<= 1)
        rmax = fmaxf(rmax, __shfl_xor(rmax, msk, 32));

      const float mn   = fmaxf(mrow[r], rmax);
      const float corr = __expf(mrow[r] - mn);
      const float p0   = __expf(a0 - mn);
      const float p1   = __expf(a1 - mn);
      float rs = p0 + p1;
#pragma unroll
      for (int msk = 1; msk <= 8; msk <<= 1)
        rs += __shfl_xor(rs, msk, 32);

      lrow[r] = lrow[r] * corr + rs;
      mrow[r] = mn;
#pragma unroll
      for (int nt = 0; nt < 4; ++nt) o[nt][r] *= corr;

      smem[w][r + half * 8][ln]      = (bf16_t)p0;
      smem[w][r + half * 8][16 + ln] = (bf16_t)p1;
    }
    __syncthreads();

    // Reload P as an A-layout 16x32 fragment (ds_load_b128 x2).
    v16bf pf;
    {
      FragU u;
      const bf16_t* pp = &smem[w][ln][half << 3];
      u.h[0] = *(const v8bf*)(pp);
      u.h[1] = *(const v8bf*)(pp + 16);
      pf = u.v;
    }

    // O += P V  (V transposed: row = hd, contiguous keys).
#pragma unroll
    for (int nt = 0; nt < 4; ++nt) {
      v16bf vf = load_frag16x32(Vbase + (nt * 16) * TT + kb, TT, 0);
      o[nt] = wmma_bf16(pf, vf, o[nt]);
    }
    __syncthreads();
  }

  // Epilogue: normalize and scatter to [b][t][h*64+hd] bf16.
  const int b = bh >> 3;
  const int h = bh & (HH - 1);
#pragma unroll
  for (int nt = 0; nt < 4; ++nt) {
#pragma unroll
    for (int r = 0; r < 8; ++r) {
      const int t = qw + r + half * 8;
      const int d = h * HDD + nt * 16 + ln;
      const float val = o[nt][r] / lrow[r];
      AO[(b * TT + t) * DD + d] = (bf16_t)val;
    }
  }
}

// ----------------------------------------------------------- k3: out GEMM
// M = 65536, N = 512, K = 512; f32 stores to d_out.
__global__ __launch_bounds__(256, 2)
void out_gemm_kernel(const bf16_t* __restrict__ AO,
                     const bf16_t* __restrict__ Wb,
                     float* __restrict__ out) {
  const int w    = threadIdx.x >> 5;
  const int wm   = w & 3;
  const int wn   = w >> 2;
  const int m0   = blockIdx.x * 256 + wm * 64;
  const int n0   = blockIdx.y * 64  + wn * 32;
  const int lane = threadIdx.x & 31;
  const int half = lane >> 4;
  const int ln   = lane & 15;

  v8f acc[4][2] = {};

  for (int k0 = 0; k0 < DD; k0 += 32) {
    __builtin_prefetch(AO + m0 * DD + ((k0 + 32) & (DD - 1)), 0, 3);
    v16bf a[4], b[2];
#pragma unroll
    for (int i = 0; i < 4; ++i)
      a[i] = load_frag16x32(AO + (m0 + i * 16) * DD, DD, k0);
#pragma unroll
    for (int j = 0; j < 2; ++j)
      b[j] = load_frag16x32(Wb + (n0 + j * 16) * DD, DD, k0);
#pragma unroll
    for (int i = 0; i < 4; ++i)
#pragma unroll
      for (int j = 0; j < 2; ++j)
        acc[i][j] = wmma_bf16(a[i], b[j], acc[i][j]);
  }

#pragma unroll
  for (int i = 0; i < 4; ++i) {
#pragma unroll
    for (int j = 0; j < 2; ++j) {
      const int n = n0 + j * 16 + ln;
#pragma unroll
      for (int r = 0; r < 8; ++r) {
        const int m = m0 + i * 16 + r + half * 8;
        out[m * DD + n] = acc[i][j][r];
      }
    }
  }
}

// ------------------------------------------------------------------- launch
extern "C" void kernel_launch(void* const* d_in, const int* in_sizes, int n_in,
                              void* d_out, int out_size, void* d_ws, size_t ws_size,
                              hipStream_t stream) {
  (void)in_sizes; (void)n_in; (void)out_size; (void)ws_size;
  const float* x    = (const float*)d_in[0];   // (B,T,D)
  const float* cosT = (const float*)d_in[1];   // (T,HD)
  const float* sinT = (const float*)d_in[2];   // (T,HD)
  const float* Wqkv = (const float*)d_in[3];   // (3D,D)
  const float* Wout = (const float*)d_in[4];   // (D,D)
  float* out = (float*)d_out;

  const size_t nX    = (size_t)BB * TT * DD;       // 33,554,432
  const size_t nWqkv = (size_t)3 * DD * DD;        //    786,432
  const size_t nWout = (size_t)DD * DD;            //    262,144
  const size_t nQKV  = (size_t)BB * HH * TT * HDD; // 33,554,432

  char* ws = (char*)d_ws;
  size_t off = 0;
  bf16_t* Xb    = (bf16_t*)(ws + off); off += nX    * sizeof(bf16_t);
  bf16_t* Wqkvb = (bf16_t*)(ws + off); off += nWqkv * sizeof(bf16_t);
  bf16_t* Woutb = (bf16_t*)(ws + off); off += nWout * sizeof(bf16_t);
  bf16_t* Qb    = (bf16_t*)(ws + off); off += nQKV  * sizeof(bf16_t);
  bf16_t* Kb    = (bf16_t*)(ws + off); off += nQKV  * sizeof(bf16_t);
  bf16_t* Vt    = (bf16_t*)(ws + off); off += nQKV  * sizeof(bf16_t);
  bf16_t* AO    = (bf16_t*)(ws + off); off += nX    * sizeof(bf16_t);

  // k0: stage to bf16
  cvt_f32_bf16_kernel<<<2048, 256, 0, stream>>>(x,    Xb,    (int)nX);
  cvt_f32_bf16_kernel<<<512,  256, 0, stream>>>(Wqkv, Wqkvb, (int)nWqkv);
  cvt_f32_bf16_kernel<<<256,  256, 0, stream>>>(Wout, Woutb, (int)nWout);

  // k1: QKV projection + RoPE (grid: M/256 x N/64)
  qkv_rope_kernel<<<dim3(256, 24), 256, 0, stream>>>(Xb, Wqkvb, cosT, sinT,
                                                     Qb, Kb, Vt);

  // k2: attention (one block per (b,h,qblock64))
  attn_kernel<<<BB * HH * (TT / 64), 128, 0, stream>>>(Qb, Kb, Vt, AO);

  // k3: output projection (grid: M/256 x N/64)
  out_gemm_kernel<<<dim3(256, 8), 256, 0, stream>>>(AO, Woutb, out);
}
